// PairwiseFunction_18124761989528
// MI455X (gfx1250) — matrix operand order repaired and checked by reading, not compiled, not run on hardware
//
#include <hip/hip_runtime.h>
#include <hip/hip_bf16.h>

// ---------------------------------------------------------------------------
// Fused edge-MLP + scatter-add for gfx1250 (MI455X), wave32 + WMMA f16.
//   h  = softplus([x[row]|x[col]] @ W0 + b0)
//   h  = softplus(h @ W1 + b1)
//   h  = h @ W2 + b2
//   out = segment_sum(h, row)   (via global_atomic_add_f32)
// ---------------------------------------------------------------------------

typedef __attribute__((ext_vector_type(16))) _Float16 v16h;
typedef __attribute__((ext_vector_type(8)))  _Float16 v8h;
typedef __attribute__((ext_vector_type(4)))  _Float16 v4h;
typedef __attribute__((ext_vector_type(8)))  float    v8f;

#define DIM_IN   64
#define DIM_HID  128
#define DIM_OUT  64
#define STRIDE   136            // padded halfs per row (272 B -> 4-bank rotation)
#define STAGE_H  (16 * STRIDE)  // one 16-row staging tile
#define WAVES    4              // 128-thread block

#define LOG2E 1.44269504088896340736f
#define LN2   0.69314718055994530942f

__device__ __forceinline__ float softplusf(float x) {
    // stable softplus on raw HW transcendentals:
    //   relu(x) + ln2 * log2(1 + exp2(-|x| * log2e))
    // argument of log2 is in [1,2] -> no denorm/range fixup needed.
    float ax   = __builtin_fabsf(x);
    float t    = __builtin_amdgcn_exp2f(-ax * LOG2E);          // v_exp_f32
    float l    = __builtin_amdgcn_logf(1.0f + t) * LN2;        // v_log_f32
    float relu = __builtin_amdgcn_fmed3f(x, 0.0f, __builtin_inff());
    return relu + l;
}

// A fragment (16x32 f16, M x K): lane m = lane%16.
// halfs 0..7  -> k = 32t + 8*hi + 0..7
// halfs 8..15 -> k = 32t + 8*hi + 16..23
__device__ __forceinline__ v16h load_a_frag(const _Float16* rowp, int t, int hi) {
    const _Float16* p = rowp + t * 32 + hi * 8;
    v8h lo = *reinterpret_cast<const v8h*>(p);
    v8h hh = *reinterpret_cast<const v8h*>(p + 16);
    v16h r;
#pragma unroll
    for (int i = 0; i < 8; ++i) { r[i] = lo[i]; r[i + 8] = hh[i]; }
    return r;
}

// B fragment (32x16 f16, K x N): lane n = lane%16.
// halfs 0..7 -> k = 32t + 8*hi + 0..7, halfs 8..15 -> k = 32t + 8*hi + 16..23.
// Weights stored n-major in LDS -> two 16B reads per lane (halfs 0..7 and 16..23 of the 32-k slice).
__device__ __forceinline__ v16h load_b_frag(const _Float16* rowp, int t, int hi) {
    const _Float16* p = rowp + t * 32 + hi * 8;
    v8h lo = *reinterpret_cast<const v8h*>(p);
    v8h hh = *reinterpret_cast<const v8h*>(p + 16);
    v16h r;
#pragma unroll
    for (int i = 0; i < 8; ++i) { r[i] = lo[i]; r[i + 8] = hh[i]; }
    return r;
}

__device__ __forceinline__ void lds_phase_fence() {
    // intra-wave: make prior DS stores visible to subsequent DS loads
    asm volatile("s_wait_dscnt 0x0" ::: "memory");
    __builtin_amdgcn_wave_barrier();
}

__global__ void __launch_bounds__(128)
zero_f32_kernel(float* __restrict__ p, int n) {
    int i = blockIdx.x * blockDim.x + threadIdx.x;
    if (i < n) p[i] = 0.0f;
}

__global__ void __launch_bounds__(128)
edge_mlp_kernel(const float* __restrict__ x,
                const int*   __restrict__ rowIdx,
                const int*   __restrict__ colIdx,
                const float* __restrict__ W0, const float* __restrict__ b0,
                const float* __restrict__ W1, const float* __restrict__ b1,
                const float* __restrict__ W2, const float* __restrict__ b2,
                float* __restrict__ out, int nEdges) {
    // ---- LDS: f16 weights (n-major, padded) + per-wave double staging ----
    __shared__ _Float16 sW0[DIM_HID * STRIDE];           // Wt0[n][k], 34 KB
    __shared__ _Float16 sW1[DIM_HID * STRIDE];           // Wt1[n][k], 34 KB
    __shared__ _Float16 sW2[DIM_OUT * STRIDE];           // Wt2[n][k], 17 KB
    __shared__ _Float16 sStage[WAVES][2][STAGE_H];       // 34 KB

    const int tid    = threadIdx.x;
    const int wave   = tid >> 5;
    const int lane   = tid & 31;
    const int lane15 = lane & 15;
    const int hi     = lane >> 4;   // 0 or 1

    // ---- one-time cooperative weight convert+transpose (f32 -> f16) ----
    for (int i = tid; i < DIM_HID * DIM_HID; i += blockDim.x) {
        int k = i >> 7, n = i & (DIM_HID - 1);
        sW0[n * STRIDE + k] = (_Float16)W0[i];
        sW1[n * STRIDE + k] = (_Float16)W1[i];
    }
    for (int i = tid; i < DIM_HID * DIM_OUT; i += blockDim.x) {
        int k = i >> 6, n = i & (DIM_OUT - 1);
        sW2[n * STRIDE + k] = (_Float16)W2[i];
    }
    __syncthreads();

    // ---- per-lane bias registers (bias depends only on n = 16j + lane15) --
    float b0v[8], b1v[8], b2v[4];
#pragma unroll
    for (int j = 0; j < 8; ++j) {
        b0v[j] = b0[j * 16 + lane15];
        b1v[j] = b1[j * 16 + lane15];
    }
#pragma unroll
    for (int j = 0; j < 4; ++j) b2v[j] = b2[j * 16 + lane15];

    _Float16* stA = sStage[wave][0];
    _Float16* stB = sStage[wave][1];

    const int nTiles   = (nEdges + 15) >> 4;
    const int tileStep = gridDim.x * WAVES;
    const char* xbytes = reinterpret_cast<const char*>(x);

    for (int tile = blockIdx.x * WAVES + wave; tile < nTiles;
         tile += tileStep) {
        const int ebase = tile << 4;

        // ---- gather + concat + f32->f16 into staging A (16 x 128) ----
        // lane<16 covers x[row] floats 4L..4L+3 ; lane>=16 covers x[col].
#pragma unroll 4
        for (int e = 0; e < 16; ++e) {
            int idx  = ebase + e;
            if (idx >= nEdges) idx = nEdges - 1;   // clamp (values unused OOB)
            int node = hi ? colIdx[idx] : rowIdx[idx];
            const float4 v =
                reinterpret_cast<const float4*>(x + (size_t)node * DIM_IN)[lane15];
            v4h h;
            h[0] = (_Float16)v.x; h[1] = (_Float16)v.y;
            h[2] = (_Float16)v.z; h[3] = (_Float16)v.w;
            *reinterpret_cast<v4h*>(stA + e * STRIDE + hi * DIM_IN + lane15 * 4) = h;
        }
        lds_phase_fence();

        // ---- prefetch next tile's node rows (overlaps the WMMA body) ----
        // 32-bit byte offsets (node*256 + lane15*16 < 2^31) keep address math
        // to one VALU op and let codegen use saddr+voffset addressing.
        {
            int nb = ebase + tileStep * 16;
            if (nb + 15 < nEdges) {
#pragma unroll 4
                for (int e = 0; e < 16; ++e) {
                    unsigned node = (unsigned)(hi ? colIdx[nb + e] : rowIdx[nb + e]);
                    unsigned off  = node * (unsigned)(DIM_IN * 4)
                                  + (unsigned)(lane15 * 16);
                    __builtin_prefetch(xbytes + off, 0, 3);
                }
            }
        }

        // ---- layer 0: stA(16x128) @ W0 -> softplus -> stB(16x128) ----
        {
            const _Float16* arow = stA + lane15 * STRIDE;
            v16h a0 = load_a_frag(arow, 0, hi);   // A tile cached in registers,
            v16h a1 = load_a_frag(arow, 1, hi);   // reused across all 8 N-tiles
            v16h a2 = load_a_frag(arow, 2, hi);
            v16h a3 = load_a_frag(arow, 3, hi);
#pragma unroll
            for (int j = 0; j < 8; ++j) {
                v8f acc;   // seed C with bias (uniform across a lane's 8 slots)
#pragma unroll
                for (int i = 0; i < 8; ++i) acc[i] = b0v[j];
                const _Float16* brow = sW0 + (j * 16 + lane15) * STRIDE;
                acc = __builtin_amdgcn_wmma_f32_16x16x32_f16(
                          false, a0, false, load_b_frag(brow, 0, hi), (short)0, acc, false, false);
                acc = __builtin_amdgcn_wmma_f32_16x16x32_f16(
                          false, a1, false, load_b_frag(brow, 1, hi), (short)0, acc, false, false);
                acc = __builtin_amdgcn_wmma_f32_16x16x32_f16(
                          false, a2, false, load_b_frag(brow, 2, hi), (short)0, acc, false, false);
                acc = __builtin_amdgcn_wmma_f32_16x16x32_f16(
                          false, a3, false, load_b_frag(brow, 3, hi), (short)0, acc, false, false);
#pragma unroll
                for (int r = 0; r < 8; ++r) {
                    float v = softplusf(acc[r]);
                    stB[(r + 8 * hi) * STRIDE + j * 16 + lane15] = (_Float16)v;
                }
            }
        }
        lds_phase_fence();

        // ---- layer 1: stB(16x128) @ W1 -> softplus -> stA(16x128) ----
        {
            const _Float16* arow = stB + lane15 * STRIDE;
            v16h a0 = load_a_frag(arow, 0, hi);
            v16h a1 = load_a_frag(arow, 1, hi);
            v16h a2 = load_a_frag(arow, 2, hi);
            v16h a3 = load_a_frag(arow, 3, hi);
#pragma unroll
            for (int j = 0; j < 8; ++j) {
                v8f acc;
#pragma unroll
                for (int i = 0; i < 8; ++i) acc[i] = b1v[j];
                const _Float16* brow = sW1 + (j * 16 + lane15) * STRIDE;
                acc = __builtin_amdgcn_wmma_f32_16x16x32_f16(
                          false, a0, false, load_b_frag(brow, 0, hi), (short)0, acc, false, false);
                acc = __builtin_amdgcn_wmma_f32_16x16x32_f16(
                          false, a1, false, load_b_frag(brow, 1, hi), (short)0, acc, false, false);
                acc = __builtin_amdgcn_wmma_f32_16x16x32_f16(
                          false, a2, false, load_b_frag(brow, 2, hi), (short)0, acc, false, false);
                acc = __builtin_amdgcn_wmma_f32_16x16x32_f16(
                          false, a3, false, load_b_frag(brow, 3, hi), (short)0, acc, false, false);
#pragma unroll
                for (int r = 0; r < 8; ++r) {
                    float v = softplusf(acc[r]);
                    stA[(r + 8 * hi) * STRIDE + j * 16 + lane15] = (_Float16)v;
                }
            }
        }
        lds_phase_fence();

        // ---- layer 2: stA(16x128) @ W2 + b2, scatter-add to out[row] ----
        {
            int rowm[8];
#pragma unroll
            for (int r = 0; r < 8; ++r) {
                int m = r + 8 * hi;
                rowm[r] = (ebase + m < nEdges) ? rowIdx[ebase + m] : -1;
            }
            const _Float16* arow = stA + lane15 * STRIDE;
            v16h a0 = load_a_frag(arow, 0, hi);
            v16h a1 = load_a_frag(arow, 1, hi);
            v16h a2 = load_a_frag(arow, 2, hi);
            v16h a3 = load_a_frag(arow, 3, hi);
#pragma unroll
            for (int j = 0; j < 4; ++j) {
                v8f acc;
#pragma unroll
                for (int i = 0; i < 8; ++i) acc[i] = b2v[j];
                const _Float16* brow = sW2 + (j * 16 + lane15) * STRIDE;
                acc = __builtin_amdgcn_wmma_f32_16x16x32_f16(
                          false, a0, false, load_b_frag(brow, 0, hi), (short)0, acc, false, false);
                acc = __builtin_amdgcn_wmma_f32_16x16x32_f16(
                          false, a1, false, load_b_frag(brow, 1, hi), (short)0, acc, false, false);
                acc = __builtin_amdgcn_wmma_f32_16x16x32_f16(
                          false, a2, false, load_b_frag(brow, 2, hi), (short)0, acc, false, false);
                acc = __builtin_amdgcn_wmma_f32_16x16x32_f16(
                          false, a3, false, load_b_frag(brow, 3, hi), (short)0, acc, false, false);
#pragma unroll
                for (int r = 0; r < 8; ++r) {
                    if (rowm[r] >= 0) {
                        float* dst = out + (size_t)rowm[r] * DIM_OUT + j * 16 + lane15;
                        __hip_atomic_fetch_add(dst, acc[r],
                                               __ATOMIC_RELAXED,
                                               __HIP_MEMORY_SCOPE_AGENT);
                    }
                }
            }
        }
        lds_phase_fence();   // stA reads done before next tile's gather writes
    }
}

extern "C" void kernel_launch(void* const* d_in, const int* in_sizes, int n_in,
                              void* d_out, int out_size, void* d_ws, size_t ws_size,
                              hipStream_t stream) {
    (void)n_in; (void)d_ws; (void)ws_size;
    const float* x   = (const float*)d_in[0];
    const int*   eix = (const int*)  d_in[1];   // [2, E]: row then col
    const float* W0  = (const float*)d_in[2];
    const float* b0  = (const float*)d_in[3];
    const float* W1  = (const float*)d_in[4];
    const float* b1  = (const float*)d_in[5];
    const float* W2  = (const float*)d_in[6];
    const float* b2  = (const float*)d_in[7];
    float*       out = (float*)d_out;

    const int E = in_sizes[1] / 2;

    // zero the scatter-add target (harness poisons d_out)
    int zb = (out_size + 127) / 128;
    zero_f32_kernel<<<zb, 128, 0, stream>>>(out, out_size);

    const int nTiles = (E + 15) / 16;
    int blocks = (nTiles + WAVES - 1) / WAVES;
    if (blocks > 1280) blocks = 1280;   // persistent-ish: amortize weight loads
    edge_mlp_kernel<<<blocks, 128, 0, stream>>>(
        x, eix, eix + E, W0, b0, W1, b1, W2, b2, out, E);
}
